// Spiking_LIF_6124623364352
// MI455X (gfx1250) — matrix operand order repaired and checked by reading, not compiled, not run on hardware
//
#include <hip/hip_runtime.h>

typedef __attribute__((ext_vector_type(16))) _Float16 v16h;
typedef __attribute__((ext_vector_type(8)))  _Float16 h8;
typedef __attribute__((ext_vector_type(4)))  _Float16 h4;
typedef __attribute__((ext_vector_type(8)))  float    v8f;
typedef __attribute__((ext_vector_type(4)))  int      i4;

#define AS_GLOBAL __attribute__((address_space(1)))
#define AS_LDS    __attribute__((address_space(3)))

#define C_IN   128
#define C_OUT  128
#define IMG_H  32
#define IMG_W  32
#define HW     1024
#define T_STEPS 8
#define BATCH  16
#define FRAMES (BATCH * T_STEPS)

#define TILE_M 64
#define TILE_N 64
#define TILE_K 64
#define LDS_STRIDE 72   // halfs per row: 144B = 9x16B, keeps b128 alignment

// CDNA5 async global->LDS path (ASYNCcnt), guarded so compile never breaks.
#if defined(__gfx1250__) &&                                                   \
    __has_builtin(__builtin_amdgcn_global_load_async_to_lds_b128) &&          \
    __has_builtin(__builtin_amdgcn_s_wait_asynccnt)
#define USE_ASYNC 1
#else
#define USE_ASYNC 0
#endif

// ---------------------------------------------------------------------------
// Pre-pass 0a: W [co][ci][3][3] f32  ->  Wh [tap][co][ci] f16 (contiguous ci)
// ---------------------------------------------------------------------------
__global__ __launch_bounds__(256) void cvt_w_kernel(
    const float* __restrict__ w, _Float16* __restrict__ wh)
{
  const int id = blockIdx.x * 256 + threadIdx.x;   // 0 .. 147455
  const int co = id / (C_IN * 9);
  const int r  = id % (C_IN * 9);
  const int ci = r / 9;
  const int kk = r % 9;
  wh[((size_t)kk * C_OUT + co) * C_IN + ci] = (_Float16)w[id];
}

// ---------------------------------------------------------------------------
// Pre-pass 0b: x f32 -> f16, same layout. 4 elements per thread.
// ---------------------------------------------------------------------------
__global__ __launch_bounds__(256) void cvt_x_kernel(
    const float4* __restrict__ x4, _Float16* __restrict__ xh)
{
  const int id = blockIdx.x * 256 + threadIdx.x;   // 0 .. 4194303
  const float4 v = x4[id];
  h4 o;
  o[0] = (_Float16)v.x; o[1] = (_Float16)v.y;
  o[2] = (_Float16)v.z; o[3] = (_Float16)v.w;
  *reinterpret_cast<h4*>(&xh[(size_t)id * 4]) = o;
}

// ---------------------------------------------------------------------------
// Kernel 1: implicit-GEMM 3x3 conv via v_wmma_f32_16x16x32_f16.
// Grid: (16 px tiles, 2 co tiles, 128 frames), 256 threads (8 waves).
// 64(co) x 64(px) f32 tile; K = 9 taps x 128 ci, staged 64 K at a time
// (18 stages, 4 WMMA per wave per stage). A staged by async LDS DMA.
// ---------------------------------------------------------------------------
__global__ __launch_bounds__(256) void conv_wmma_kernel(
    const _Float16* __restrict__ xh, const _Float16* __restrict__ wh,
    float* __restrict__ y)
{
  __shared__ __align__(16) _Float16 As[TILE_M][LDS_STRIDE];  // [co][k]
  __shared__ __align__(16) _Float16 Bs[TILE_N][LDS_STRIDE];  // [px][k]

  const int tid   = threadIdx.x;
  const int lane  = tid & 31;
  const int wid   = tid >> 5;
  const int mi    = wid & 3;     // co sub-tile (0..3)
  const int ni    = wid >> 2;    // pixel pair sub-tile (0..1)
  const int halfl = lane >> 4;   // 0 | 1 (wave half)
  const int l16   = lane & 15;

  const int ptile   = blockIdx.x * TILE_N;
  const int co_base = blockIdx.y * TILE_M;
  const int frame   = blockIdx.z;

  const _Float16* __restrict__ xf = xh + (size_t)frame * C_IN * HW;

  v8f acc0 = {};
  v8f acc1 = {};

  for (int kk = 0; kk < 9; ++kk) {
    const int dh = kk / 3 - 1;
    const int dw = kk % 3 - 1;
    #pragma unroll
    for (int cc = 0; cc < 2; ++cc) {
      const int ci0 = cc * TILE_K;

      // ---- stage A: Wh[kk][co_base+row][ci0 + 0..63], 64x64 halfs ----
      // two 16B chunks per thread, fully coalesced
#if USE_ASYNC
      {
        #pragma unroll
        for (int i = 0; i < 2; ++i) {
          const int c   = tid + i * 256;
          const int row = c >> 3;      // 0..63
          const int kb  = c & 7;       // 0..7 (x8 halfs)
          const _Float16* g =
              &wh[((size_t)kk * C_OUT + co_base + row) * C_IN + ci0 + kb * 8];
          __builtin_amdgcn_global_load_async_to_lds_b128(
              (AS_GLOBAL i4*)g,
              (AS_LDS i4*)&As[row][kb * 8],
              0, 0);
        }
      }
#else
      {
        #pragma unroll
        for (int i = 0; i < 2; ++i) {
          const int c   = tid + i * 256;
          const int row = c >> 3;
          const int kb  = c & 7;
          const h8 v = *reinterpret_cast<const h8*>(
              &wh[((size_t)kk * C_OUT + co_base + row) * C_IN + ci0 + kb * 8]);
          *reinterpret_cast<h8*>(&As[row][kb * 8]) = v;
        }
      }
#endif

      // ---- stage B: im2col xh[ci0+k][h+dh][w+dw], 64x64 halfs ([n][k]) ----
      {
        const int n    = tid & 63;
        const int k0   = tid >> 6;  // 0..3
        const int p    = ptile + n;
        const int h    = (p >> 5) + dh;
        const int wcol = (p & 31) + dw;
        const bool inb = (h >= 0) & (h < IMG_H) & (wcol >= 0) & (wcol < IMG_W);
        const size_t base = (size_t)h * IMG_W + wcol;
        #pragma unroll
        for (int i = 0; i < 16; ++i) {
          const int k = k0 + i * 4;
          const _Float16 v =
              inb ? xf[(size_t)(ci0 + k) * HW + base] : (_Float16)0.0f;
          Bs[n][k] = v;
        }
      }

#if USE_ASYNC
      __builtin_amdgcn_s_wait_asynccnt(0);
#endif
      __syncthreads();

      // ---- A fragments (16x32 f16 each, K-low / K-high) ----
      const int m_lane = mi * 16 + l16;
      v16h a0, a1;
      {
        const h8 lo0 = *reinterpret_cast<const h8*>(&As[m_lane][halfl * 16]);
        const h8 hi0 = *reinterpret_cast<const h8*>(&As[m_lane][halfl * 16 + 8]);
        const h8 lo1 = *reinterpret_cast<const h8*>(&As[m_lane][32 + halfl * 16]);
        const h8 hi1 = *reinterpret_cast<const h8*>(&As[m_lane][40 + halfl * 16]);
        #pragma unroll
        for (int i = 0; i < 8; ++i) {
          a0[i] = lo0[i]; a0[i + 8] = hi0[i];
          a1[i] = lo1[i]; a1[i + 8] = hi1[i];
        }
      }

      // ---- B fragments (32x16 f16), two n-tiles x two K-halves ----
      const int n0 = ni * 32 + l16;
      const int n1 = n0 + 16;
      v16h b00, b01, b10, b11;   // b{ntile}{khalf}
      {
        const h8 p00a = *reinterpret_cast<const h8*>(&Bs[n0][halfl * 16]);
        const h8 p00b = *reinterpret_cast<const h8*>(&Bs[n0][halfl * 16 + 8]);
        const h8 p01a = *reinterpret_cast<const h8*>(&Bs[n0][32 + halfl * 16]);
        const h8 p01b = *reinterpret_cast<const h8*>(&Bs[n0][40 + halfl * 16]);
        const h8 p10a = *reinterpret_cast<const h8*>(&Bs[n1][halfl * 16]);
        const h8 p10b = *reinterpret_cast<const h8*>(&Bs[n1][halfl * 16 + 8]);
        const h8 p11a = *reinterpret_cast<const h8*>(&Bs[n1][32 + halfl * 16]);
        const h8 p11b = *reinterpret_cast<const h8*>(&Bs[n1][40 + halfl * 16]);
        #pragma unroll
        for (int i = 0; i < 8; ++i) {
          b00[i] = p00a[i]; b00[i + 8] = p00b[i];
          b01[i] = p01a[i]; b01[i + 8] = p01b[i];
          b10[i] = p10a[i]; b10[i + 8] = p10b[i];
          b11[i] = p11a[i]; b11[i + 8] = p11b[i];
        }
      }

      acc0 = __builtin_amdgcn_wmma_f32_16x16x32_f16(
          false, a0, false, b00, (short)0, acc0, false, false);
      acc1 = __builtin_amdgcn_wmma_f32_16x16x32_f16(
          false, a0, false, b10, (short)0, acc1, false, false);
      acc0 = __builtin_amdgcn_wmma_f32_16x16x32_f16(
          false, a1, false, b01, (short)0, acc0, false, false);
      acc1 = __builtin_amdgcn_wmma_f32_16x16x32_f16(
          false, a1, false, b11, (short)0, acc1, false, false);

      __syncthreads();
    }
  }

  // ---- epilogue: spill f32 accumulators, C/D layout M = r + 8*half ----
  float* __restrict__ yf = y + (size_t)frame * C_OUT * HW;
  #pragma unroll
  for (int r = 0; r < 8; ++r) {
    const int co = co_base + mi * 16 + halfl * 8 + r;
    yf[(size_t)co * HW + ptile + ni * 32 + l16]      = acc0[r];
    yf[(size_t)co * HW + ptile + ni * 32 + 16 + l16] = acc1[r];
  }
}

// ---------------------------------------------------------------------------
// Kernel 2: LIF temporal scan. One thread per (b, co, h, w) site; serial over
// T=8, fully coalesced in the pixel dimension at each timestep.
// ---------------------------------------------------------------------------
__global__ __launch_bounds__(256) void lif_scan_kernel(
    const float* __restrict__ y, const float* __restrict__ bias,
    const float* __restrict__ thr_p, float* __restrict__ out)
{
  const int idx = blockIdx.x * blockDim.x + threadIdx.x;  // 0 .. 2^21-1
  const int p  = idx & 1023;          // pixel
  const int co = (idx >> 10) & 127;   // channel
  const int b  = idx >> 17;           // batch (Co*H*W = 2^17)

  const float th    = thr_p[0];
  const float bi    = bias[co];
  const float decay = 1.0f - 1.0f / 20.0f;  // tau = 20

  float m = 0.5f * th;
  float s = 0.0f;

  #pragma unroll
  for (int t = 0; t < T_STEPS; ++t) {
    const size_t off =
        ((size_t)(b * T_STEPS + t) * C_OUT + co) * (size_t)HW + p;
    m = decay * m + (y[off] + bi);
    const float spk = (m >= th) ? 1.0f : 0.0f;   // fire
    m -= spk * th;                               // soft reset
    s += spk;
    const float inh = ((m <= -0.001f) && (s > 0.0f)) ? 1.0f : 0.0f;
    m += inh * th;                               // inhibit undo
    s -= inh;
    out[off] = (spk - inh) * th;
  }
}

// ---------------------------------------------------------------------------
extern "C" void kernel_launch(void* const* d_in, const int* in_sizes, int n_in,
                              void* d_out, int out_size, void* d_ws,
                              size_t ws_size, hipStream_t stream) {
  const float* x   = (const float*)d_in[0];  // [16, 8, 128, 32, 32]
  const float* w   = (const float*)d_in[1];  // [128, 128, 3, 3]
  const float* b   = (const float*)d_in[2];  // [128]
  const float* thr = (const float*)d_in[3];  // scalar
  float* out = (float*)d_out;                // [16, 8, 128, 32, 32]

  // workspace carve-up (<= ~101.5 MB):
  //   y  : f32 conv output, 64 MB
  //   Wh : f16 reordered weights, 288 KB (at +64 MB, padded to 512 KB)
  //   xh : f16 input, 32 MB
  char* wsc = (char*)d_ws;
  float*    y  = (float*)wsc;
  _Float16* wh = (_Float16*)(wsc + (size_t)67108864);
  _Float16* xh = (_Float16*)(wsc + (size_t)67108864 + 524288);

  cvt_w_kernel<<<(C_OUT * C_IN * 9) / 256, 256, 0, stream>>>(w, wh);

  const int x4n = (BATCH * T_STEPS * C_IN * HW) / 4;   // 4,194,304
  cvt_x_kernel<<<x4n / 256, 256, 0, stream>>>((const float4*)x, xh);

  dim3 g1(HW / TILE_N, C_OUT / TILE_M, FRAMES);        // (16, 2, 128)
  conv_wmma_kernel<<<g1, 256, 0, stream>>>(xh, wh, y);

  const int sites = BATCH * C_OUT * HW;                // 2,097,152
  lif_scan_kernel<<<sites / 256, 256, 0, stream>>>(y, b, thr, out);
}
